// MultiHopGATLayer_66288525247052
// MI455X (gfx1250) — matrix operand order repaired and checked by reading, not compiled, not run on hardware
//
#include <hip/hip_runtime.h>

#define N_NODES 50000
#define N_EDGES 800000
#define E_TOT   (N_EDGES + N_NODES)
#define HEADS   8
#define HDIM    16
#define OUTC    128
#define NEG_SLOPE 0.2f
#define BN_EPS  1e-5f

typedef __attribute__((ext_vector_type(16))) __bf16       bf16x16;
typedef __attribute__((ext_vector_type(8)))  float        floatx8;
typedef __attribute__((ext_vector_type(4)))  unsigned int uint4v;

union Frag { bf16x16 v; uint4v q[2]; };

__device__ __forceinline__ unsigned short f2bf(float f) {
    unsigned u = __float_as_uint(f);
    unsigned r = 0x7FFFu + ((u >> 16) & 1u);      // round-to-nearest-even
    return (unsigned short)((u + r) >> 16);
}
__device__ __forceinline__ unsigned fkey(float f) {   // order-preserving float->uint
    unsigned u = __float_as_uint(f);
    return (u & 0x80000000u) ? ~u : (u | 0x80000000u);
}
__device__ __forceinline__ float kinv(unsigned k) {
    unsigned u = (k & 0x80000000u) ? (k & 0x7FFFFFFFu) : ~k;
    return __uint_as_float(u);
}
__device__ __forceinline__ float lrelu(float v) { return v > 0.f ? v : NEG_SLOPE * v; }

// ---------------- GEMM: h = x @ W via v_wmma_f32_16x16x32_bf16; also zeroes out ----
#define LDA 72   // 64 + 8 pad shorts: conflict-free b128 LDS reads

__global__ void k_gemm(const float* __restrict__ x, const float* __restrict__ W,
                       float* __restrict__ h, float* __restrict__ out) {
    __shared__ unsigned short ldsA[128 * LDA];   // x tile, bf16, [row][k]
    __shared__ unsigned short ldsB[128 * LDA];   // W^T tile, bf16, [n][k]
    const int tid  = threadIdx.x;
    const int lane = tid & 31, wave = tid >> 5;
    const int r = lane & 15, half = lane >> 4;
    const int row0 = blockIdx.x * 128;

    floatx8 acc[8];
    floatx8 zero = {};
#pragma unroll
    for (int i = 0; i < 8; ++i) acc[i] = zero;

    for (int kc = 0; kc < 128; kc += 64) {
        // stage x[row0..row0+127, kc..kc+63] as bf16
#pragma unroll
        for (int i = 0; i < 8; ++i) {
            int idx = tid + i * 256;              // 0..2047
            int rr  = idx >> 4;
            int c4  = (idx & 15) << 2;
            int gr  = row0 + rr;
            float4 v = make_float4(0.f, 0.f, 0.f, 0.f);
            if (gr < N_NODES) v = *(const float4*)(&x[(long)gr * 128 + kc + c4]);
            unsigned short* p = &ldsA[rr * LDA + c4];
            p[0] = f2bf(v.x); p[1] = f2bf(v.y); p[2] = f2bf(v.z); p[3] = f2bf(v.w);
        }
        // stage W^T: ldsB[n][kl] = W[kc+kl][n]
#pragma unroll
        for (int i = 0; i < 32; ++i) {
            int idx = tid + i * 256;              // 0..8191
            int n   = idx & 127;
            int kl  = idx >> 7;
            ldsB[n * LDA + kl] = f2bf(W[(long)(kc + kl) * 128 + n]);
        }
        __syncthreads();

#pragma unroll
        for (int kk = 0; kk < 64; kk += 32) {
            // A frag (16x32 bf16): half0: v0-3=K0-7,v4-7=K16-23; half1: +8
            Frag a;
            const unsigned short* pa = &ldsA[(wave * 16 + r) * LDA + kk + half * 8];
            a.q[0] = *(const uint4v*)(pa);
            a.q[1] = *(const uint4v*)(pa + 16);
#pragma unroll
            for (int nt = 0; nt < 8; ++nt) {
                // B frag (32x16 bf16): half0: v0-3=K0-7,v4-7=K16-23; half1: +8
                Frag b;
                const unsigned short* pb = &ldsB[(nt * 16 + r) * LDA + kk + half * 8];
                b.q[0] = *(const uint4v*)(pb);
                b.q[1] = *(const uint4v*)(pb + 16);
                acc[nt] = __builtin_amdgcn_wmma_f32_16x16x32_bf16(
                    false, a.v, false, b.v, (short)0, acc[nt], false, false);
            }
        }
        __syncthreads();
    }

    // C layout: VGPR v -> M = half*8 + v, N = r
    const int orow0 = row0 + wave * 16 + half * 8;
#pragma unroll
    for (int nt = 0; nt < 8; ++nt) {
        int col = nt * 16 + r;
#pragma unroll
        for (int v = 0; v < 8; ++v) {
            int grow = orow0 + v;
            if (grow < N_NODES) {
                h[(long)grow * 128 + col]   = acc[nt][v];
                out[(long)grow * 128 + col] = 0.0f;   // init accumulation buffer
            }
        }
    }
}

// ---------------- per-node attention halves + softmax-state init -------------------
__global__ void k_att(const float* __restrict__ h,
                      const float* __restrict__ att_src, const float* __restrict__ att_dst,
                      float* __restrict__ asrc, float* __restrict__ adst,
                      unsigned* __restrict__ mkey, float* __restrict__ denom,
                      float* __restrict__ gsum, float* __restrict__ gsumsq) {
    int gid = blockIdx.x * 256 + threadIdx.x;
    if (gid < 128) { gsum[gid] = 0.f; gsumsq[gid] = 0.f; }
    if (gid >= N_NODES * HEADS) return;
    int n = gid >> 3, hd = gid & 7;
    const float* hp = &h[(long)n * 128 + hd * HDIM];
    float s = 0.f, d = 0.f;
#pragma unroll
    for (int i = 0; i < HDIM; ++i) {
        float hv = hp[i];
        s += hv * att_src[hd * HDIM + i];
        d += hv * att_dst[hd * HDIM + i];
    }
    asrc[gid] = s; adst[gid] = d;
    mkey[gid] = 0u;        // below key of any finite float produced here
    denom[gid] = 0.f;
}

__device__ __forceinline__ void edge_nodes(const int* __restrict__ ei, int e, int& s, int& d) {
    if (e < N_EDGES) { s = ei[e]; d = ei[N_EDGES + e]; }
    else             { s = d = e - N_EDGES; }           // appended self-loops
}

// ---------------- segment max ------------------------------------------------------
__global__ void k_edge_max(const int* __restrict__ ei, const float* __restrict__ asrc,
                           const float* __restrict__ adst, unsigned* __restrict__ mkey) {
    int gid = blockIdx.x * 256 + threadIdx.x;
    if (gid >= E_TOT * HEADS) return;
    int e = gid >> 3, hd = gid & 7, s, d;
    edge_nodes(ei, e, s, d);
    float v = lrelu(asrc[s * 8 + hd] + adst[d * 8 + hd]);
    atomicMax(&mkey[d * 8 + hd], fkey(v));
}

// ---------------- segment sum of exp ----------------------------------------------
__global__ void k_edge_sum(const int* __restrict__ ei, const float* __restrict__ asrc,
                           const float* __restrict__ adst, const unsigned* __restrict__ mkey,
                           float* __restrict__ denom) {
    int gid = blockIdx.x * 256 + threadIdx.x;
    if (gid >= E_TOT * HEADS) return;
    int e = gid >> 3, hd = gid & 7, s, d;
    edge_nodes(ei, e, s, d);
    float v = lrelu(asrc[s * 8 + hd] + adst[d * 8 + hd]);
    float m = kinv(mkey[d * 8 + hd]);
    atomicAdd(&denom[d * 8 + hd], __expf(v - m));
}

// ---------------- weighted scatter-aggregate: 1 thread per (edge, channel) --------
__global__ void k_agg(const int* __restrict__ ei, const float* __restrict__ asrc,
                      const float* __restrict__ adst, const unsigned* __restrict__ mkey,
                      const float* __restrict__ denom, const float* __restrict__ h,
                      float* __restrict__ out) {
    int gid = blockIdx.x * 256 + threadIdx.x;
    if (gid >= E_TOT * OUTC) return;
    int e = gid >> 7, c = gid & 127, hd = c >> 4, s, d;
    edge_nodes(ei, e, s, d);
    float v = lrelu(asrc[s * 8 + hd] + adst[d * 8 + hd]);
    float m   = kinv(mkey[d * 8 + hd]);
    float den = denom[d * 8 + hd];
    float alpha = __expf(v - m) / (den + 1e-16f);
    atomicAdd(&out[(long)d * 128 + c], alpha * h[(long)s * 128 + c]);
}

// ---------------- BatchNorm stats --------------------------------------------------
__global__ void k_bn_stats(const float* __restrict__ out,
                           float* __restrict__ gsum, float* __restrict__ gsumsq) {
    __shared__ float ss[256], sq[256];
    int tid = threadIdx.x;
    int c = tid & 127, rsel = tid >> 7;
    long r0 = (long)blockIdx.x * 128;
    float s = 0.f, q = 0.f;
    for (int r = rsel; r < 128; r += 2) {
        long gr = r0 + r;
        if (gr < N_NODES) { float v = out[gr * 128 + c]; s += v; q += v * v; }
    }
    ss[tid] = s; sq[tid] = q;
    __syncthreads();
    if (tid < 128) {
        atomicAdd(&gsum[tid],   ss[tid] + ss[tid + 128]);
        atomicAdd(&gsumsq[tid], sq[tid] + sq[tid + 128]);
    }
}

// ---------------- BatchNorm apply + ReLU (bias cancels through BN) ----------------
__global__ void k_bn_apply(const float* __restrict__ gsum, const float* __restrict__ gsumsq,
                           const float* __restrict__ gamma, const float* __restrict__ beta,
                           float* __restrict__ out) {
    int gid = blockIdx.x * 256 + threadIdx.x;
    if (gid >= N_NODES * OUTC) return;
    int c = gid & 127;
    const float invN = 1.0f / (float)N_NODES;
    float mean = gsum[c] * invN;
    float var  = gsumsq[c] * invN - mean * mean;
    float scale = __frsqrt_rn(var + BN_EPS) * gamma[c];
    float val = (out[gid] - mean) * scale + beta[c];
    out[gid] = fmaxf(val, 0.0f);
}

extern "C" void kernel_launch(void* const* d_in, const int* in_sizes, int n_in,
                              void* d_out, int out_size, void* d_ws, size_t ws_size,
                              hipStream_t stream) {
    const float* x       = (const float*)d_in[0];
    const int*   ei      = (const int*)  d_in[1];
    const float* W       = (const float*)d_in[2];
    const float* att_src = (const float*)d_in[3];
    const float* att_dst = (const float*)d_in[4];
    // d_in[5] = bias: exactly cancels through training-mode BatchNorm, unused
    const float* gamma   = (const float*)d_in[6];
    const float* beta    = (const float*)d_in[7];
    float* out = (float*)d_out;

    float*    h      = (float*)d_ws;                       // N*128
    float*    asrc   = h      + (long)N_NODES * 128;       // N*8
    float*    adst   = asrc   + (long)N_NODES * 8;         // N*8
    unsigned* mkey   = (unsigned*)(adst + (long)N_NODES * 8); // N*8
    float*    denom  = (float*)(mkey + (long)N_NODES * 8); // N*8
    float*    gsum   = denom  + (long)N_NODES * 8;         // 128
    float*    gsumsq = gsum   + 128;                       // 128

    const int nblk_gemm = (N_NODES + 127) / 128;
    const int nblk_att  = (N_NODES * HEADS + 255) / 256;
    const int nblk_eh   = (E_TOT * HEADS + 255) / 256;
    const int nblk_agg  = (E_TOT * OUTC + 255) / 256;
    const int nblk_bn   = (N_NODES + 127) / 128;
    const int nblk_nc   = (N_NODES * OUTC + 255) / 256;

    k_gemm     <<<nblk_gemm, 256, 0, stream>>>(x, W, h, out);
    k_att      <<<nblk_att,  256, 0, stream>>>(h, att_src, att_dst, asrc, adst, mkey, denom, gsum, gsumsq);
    k_edge_max <<<nblk_eh,   256, 0, stream>>>(ei, asrc, adst, mkey);
    k_edge_sum <<<nblk_eh,   256, 0, stream>>>(ei, asrc, adst, mkey, denom);
    k_agg      <<<nblk_agg,  256, 0, stream>>>(ei, asrc, adst, mkey, denom, h, out);
    k_bn_stats <<<nblk_bn,   256, 0, stream>>>(out, gsum, gsumsq);
    k_bn_apply <<<nblk_nc,   256, 0, stream>>>(gsum, gsumsq, gamma, beta, out);
}
